// SSD_56667798504215
// MI455X (gfx1250) — matrix operand order repaired and checked by reading, not compiled, not run on hardware
//
#include <hip/hip_runtime.h>

typedef _Float16 half_t;
typedef __attribute__((ext_vector_type(16))) _Float16 v16h;
typedef __attribute__((ext_vector_type(8)))  _Float16 v8h;
typedef __attribute__((ext_vector_type(8)))  float    v8f;
typedef __attribute__((ext_vector_type(4)))  unsigned int u32x4;
typedef __attribute__((ext_vector_type(8)))  int          i32x8;
typedef __attribute__((ext_vector_type(4)))  int          i32x4;

#define NWAVE 4
#define NTHR  128

// ---------------------------------------------------------------------------
// elementwise fp32 -> fp16
// ---------------------------------------------------------------------------
__global__ void cvt_f32_f16(const float* __restrict__ in, half_t* __restrict__ out, long n) {
    long i = (long)blockIdx.x * blockDim.x + threadIdx.x;
    if (i < n) out[i] = (half_t)in[i];
}

// ---------------------------------------------------------------------------
// Implicit-GEMM convolution via V_WMMA_F32_16X16X32_F16.
//   GEMM view: M = Cout, N = B*Hout*Wout, K = Cin*KH*KW
//   Block: 128 threads (4 waves). Block tile: 64 Cout x 64 pixels, K-step 32.
//   A tile (weights, 64x32, K contiguous) is DMA'd into LDS by the Tensor
//   Data Mover (wave 0 issues TENSOR_LOAD_TO_LDS; OOB rows/cols zero-fill).
//   B tile (im2col gather, 16 pixels x 32 K per wave) is staged manually.
//   Each wave reuses its B fragment across 4 WMMAs (4 Cout sub-tiles).
// ---------------------------------------------------------------------------
__global__ __launch_bounds__(NTHR)
void conv_wmma_kernel(const half_t* __restrict__ act, const half_t* __restrict__ wgt,
                      const float* __restrict__ bias,
                      half_t* __restrict__ out_h, float* __restrict__ out_f,
                      int Bn, int Cin, int H, int W,
                      int Cout, int Hout, int Wout,
                      int KH, int KW, int stride, int pad, int dil, int relu)
{
    __shared__ __align__(16) half_t As[64 * 32];          // 64 cout rows x 32 k
    __shared__ __align__(16) half_t Bs[NWAVE][16 * 32];   // 16 pixels x 32 k per wave
    __shared__ int  sIY[NWAVE][16];
    __shared__ int  sIX[NWAVE][16];
    __shared__ long sPB[NWAVE][16];

    const int tid  = threadIdx.x;
    const int lane = tid & 31;
    const int wave = tid >> 5;

    const int P   = Bn * Hout * Wout;
    const int K   = Cin * KH * KW;
    const int khw = KH * KW;
    const int HWo = Hout * Wout;
    const int co0 = blockIdx.y * 64;
    const int p0  = blockIdx.x * (NWAVE * 16) + wave * 16;

    if (lane < 16) {                       // per-wave pixel table (16 output pixels)
        int p = p0 + lane;
        long base = -1; int iy0 = 0, ix0 = 0;
        if (p < P) {
            int b = p / HWo; int rem = p - b * HWo;
            int oy = rem / Wout; int ox = rem - oy * Wout;
            iy0 = oy * stride - pad;
            ix0 = ox * stride - pad;
            base = (long)b * Cin * H * W;
        }
        sIY[wave][lane] = iy0; sIX[wave][lane] = ix0; sPB[wave][lane] = base;
    }

    const unsigned ldsOff = (unsigned)(unsigned long long)(const void*)As;

    v8f acc0 = {}, acc1 = {}, acc2 = {}, acc3 = {};

    for (int kk = 0; kk < K; kk += 32) {
        __syncthreads();
        // ---- A tile via Tensor Data Mover: 64x32 f16 sub-tile of the OIHW
        //      weight tensor (row stride = K elems). OOB reads zero-fill.
        if (wave == 0) {
            unsigned long long gaddr = (unsigned long long)(const char*)wgt
                                     + 2ull * ((unsigned long long)(unsigned)co0 * (unsigned)K
                                               + (unsigned)kk);
            unsigned td0 = (unsigned)(K - kk);        // remaining cols from tile start
            unsigned td1 = (unsigned)(Cout - co0);    // remaining rows from tile start
            unsigned long long st0 = (unsigned long long)(unsigned)K;

            u32x4 g0;
            g0[0] = 1u;                                               // count=1
            g0[1] = ldsOff;                                           // lds_addr (bytes)
            g0[2] = (unsigned)(gaddr & 0xFFFFFFFFull);                // global_addr[31:0]
            g0[3] = (unsigned)((gaddr >> 32) & 0x01FFFFFFull)         // global_addr[56:32]
                  | (2u << 30);                                       // type=2 (image)
            i32x8 g1;
            g1[0] = (int)(1u << 16);                                  // data_size=1 (2B)
            g1[1] = (int)((td0 & 0xFFFFu) << 16);                     // tensor_dim0[15:0]
            g1[2] = (int)(((td0 >> 16) & 0xFFFFu) | ((td1 & 0xFFFFu) << 16));
            g1[3] = (int)(((td1 >> 16) & 0xFFFFu) | (32u << 16));     // tile_dim0 = 32
            g1[4] = (int)64u;                                         // tile_dim1 = 64
            g1[5] = (int)(unsigned)(st0 & 0xFFFFFFFFull);             // dim0 stride lo
            g1[6] = (int)(unsigned)((st0 >> 32) & 0xFFFFu);           // dim0 stride hi
            g1[7] = 0;
            i32x4 gz4 = {0, 0, 0, 0};
            i32x8 gz8 = {0, 0, 0, 0, 0, 0, 0, 0};
            __builtin_amdgcn_tensor_load_to_lds(g0, g1, gz4, gz4, gz8, 0);
        }
        // ---- B tile (im2col gather): this lane owns column k=lane
        {
            const int gk = kk + lane;
            int cin = 0, r = 0, s2 = 0;
            const bool kval = gk < K;
            if (kval) {
                cin = gk / khw; int rr = gk - cin * khw;
                r = rr / KW; s2 = rr - r * KW;
            }
            const long cbase = (long)cin * H * W;
            const int ry = r * dil, sx = s2 * dil;
            for (int n = 0; n < 16; ++n) {
                half_t v = (half_t)0;
                long base = sPB[wave][n];
                if (kval && base >= 0) {
                    int iy = sIY[wave][n] + ry;
                    int ix = sIX[wave][n] + sx;
                    if (iy >= 0 && iy < H && ix >= 0 && ix < W)
                        v = act[base + cbase + (long)iy * W + ix];
                }
                Bs[wave][n * 32 + lane] = v;
            }
        }
        if (wave == 0) __builtin_amdgcn_s_wait_tensorcnt(0);  // TDM done before barrier
        __syncthreads();

        // ---- B fragment once, A fragments x4, 4 WMMAs (block-uniform guards)
        const int hsel = lane >> 4;
        const int row  = lane & 15;
        v8h b0 = *(const v8h*)&Bs[wave][row * 32 + hsel * 8];
        v8h b1 = *(const v8h*)&Bs[wave][row * 32 + hsel * 8 + 16];
        v16h bv;
        #pragma unroll
        for (int e = 0; e < 8; ++e) { bv[e] = b0[e]; bv[e + 8] = b1[e]; }

        #pragma unroll
        for (int t = 0; t < 4; ++t) {
            if (co0 + 16 * t < Cout) {      // uniform across block
                int arow = (16 * t + row) * 32;
                v8h a0 = *(const v8h*)&As[arow + hsel * 8];
                v8h a1 = *(const v8h*)&As[arow + hsel * 8 + 16];
                v16h av;
                #pragma unroll
                for (int e = 0; e < 8; ++e) { av[e] = a0[e]; av[e + 8] = a1[e]; }
                v8f* pa = (t == 0) ? &acc0 : (t == 1) ? &acc1 : (t == 2) ? &acc2 : &acc3;
                *pa = __builtin_amdgcn_wmma_f32_16x16x32_f16(false, av, false, bv,
                                                             (short)0, *pa, false, false);
            }
        }
    }

    // ---- store D: N = lane&15 (pixel), M = 16*t + r + 8*(lane>>4) (cout)
    int p = p0 + (lane & 15);
    if (p < P) {
        int b = p / HWo; int rem = p - b * HWo;
        #pragma unroll
        for (int t = 0; t < 4; ++t) {
            const v8f& acc = (t == 0) ? acc0 : (t == 1) ? acc1 : (t == 2) ? acc2 : acc3;
            #pragma unroll
            for (int r = 0; r < 8; ++r) {
                int co = co0 + 16 * t + r + 8 * (lane >> 4);
                if (co < Cout) {
                    float v = acc[r] + bias[co];
                    if (relu) v = v > 0.f ? v : 0.f;
                    long o = ((long)b * Cout + co) * HWo + rem;
                    if (out_f) out_f[o] = v;
                    else       out_h[o] = (half_t)v;
                }
            }
        }
    }
}

// ---------------------------------------------------------------------------
// max pooling (trailing ceil-pad handled by bounds check; pool5 uses padT/L=1)
// ---------------------------------------------------------------------------
__global__ void maxpool_kernel(const half_t* __restrict__ in, half_t* __restrict__ out,
                               int Bn, int C, int H, int W, int Hout, int Wout,
                               int k, int s, int padT, int padL)
{
    long idx = (long)blockIdx.x * blockDim.x + threadIdx.x;
    long total = (long)Bn * C * Hout * Wout;
    if (idx >= total) return;
    int ox = (int)(idx % Wout); long t = idx / Wout;
    int oy = (int)(t % Hout);   t /= Hout;
    int c  = (int)(t % C);      int b = (int)(t / C);
    const half_t* p = in + ((long)b * C + c) * H * W;
    float m = -3.4e38f;
    for (int dy = 0; dy < k; ++dy) {
        int iy = oy * s - padT + dy;
        if (iy < 0 || iy >= H) continue;
        for (int dx = 0; dx < k; ++dx) {
            int ix = ox * s - padL + dx;
            if (ix < 0 || ix >= W) continue;
            float v = (float)p[(long)iy * W + ix];
            m = v > m ? v : m;
        }
    }
    out[idx] = (half_t)m;
}

// ---------------------------------------------------------------------------
// L2Norm across channels (conv4_3): s = x / sqrt(sum x^2 + 1e-10) * scale[c]
// ---------------------------------------------------------------------------
__global__ void l2norm_kernel(const half_t* __restrict__ in, const float* __restrict__ scale,
                              half_t* __restrict__ out, int Bn, int C, int HW)
{
    long idx = (long)blockIdx.x * blockDim.x + threadIdx.x;
    long total = (long)Bn * HW;
    if (idx >= total) return;
    int hw = (int)(idx % HW); int b = (int)(idx / HW);
    const half_t* p = in + (long)b * C * HW + hw;
    float ss = 1e-10f;
    for (int c = 0; c < C; ++c) { float v = (float)p[(long)c * HW]; ss += v * v; }
    float inv = rsqrtf(ss);
    half_t* q = out + (long)b * C * HW + hw;
    for (int c = 0; c < C; ++c)
        q[(long)c * HW] = (half_t)((float)p[(long)c * HW] * inv * scale[c]);
}

// ---------------------------------------------------------------------------
// Head scatter: NCHW head output -> [B, 8732, comp] (NHWC flatten + concat)
// ---------------------------------------------------------------------------
__global__ void scatter_head(const float* __restrict__ src, float* __restrict__ dst,
                             int Bn, int Hs, int Ws, int nb, int comp,
                             int prior_off, int total_pri)
{
    long idx = (long)blockIdx.x * blockDim.x + threadIdx.x;
    long total = (long)Bn * Hs * Ws * nb * comp;
    if (idx >= total) return;
    int c = (int)(idx % comp); long t = idx / comp;
    int a = (int)(t % nb);     t /= nb;
    int x = (int)(t % Ws);     t /= Ws;
    int y = (int)(t % Hs);     int b = (int)(t / Hs);
    int ch = a * comp + c;
    float v = src[(((long)b * (nb * comp) + ch) * Hs + y) * Ws + x];
    dst[((long)b * total_pri + prior_off + (y * Ws + x) * nb + a) * (long)comp + c] = v;
}

// ---------------------------------------------------------------------------
// 21-class softmax per prior
// ---------------------------------------------------------------------------
__global__ void softmax21_kernel(const float* __restrict__ in, float* __restrict__ out, long n)
{
    long i = (long)blockIdx.x * blockDim.x + threadIdx.x;
    if (i >= n) return;
    const float* p = in + i * 21;
    float mx = p[0];
    for (int c = 1; c < 21; ++c) mx = p[c] > mx ? p[c] : mx;
    float e[21]; float s = 0.f;
    for (int c = 0; c < 21; ++c) { e[c] = expf(p[c] - mx); s += e[c]; }
    float inv = 1.f / s;
    float* q = out + i * 21;
    for (int c = 0; c < 21; ++c) q[c] = e[c] * inv;
}

// ---------------------------------------------------------------------------
// Prior-box decode: variances (0.1, 0.2)
// ---------------------------------------------------------------------------
__global__ void decode_kernel(const float* __restrict__ loc, const float* __restrict__ priors,
                              float* __restrict__ out, int Bn, int P)
{
    long idx = (long)blockIdx.x * blockDim.x + threadIdx.x;
    long total = (long)Bn * P;
    if (idx >= total) return;
    int p = (int)(idx % P);
    const float* l  = loc + idx * 4;
    const float* pr = priors + (long)p * 4;
    float cx = pr[0] + l[0] * 0.1f * pr[2];
    float cy = pr[1] + l[1] * 0.1f * pr[3];
    float w  = pr[2] * expf(l[2] * 0.2f);
    float h  = pr[3] * expf(l[3] * 0.2f);
    float* q = out + idx * 4;
    q[0] = cx - w * 0.5f; q[1] = cy - h * 0.5f;
    q[2] = cx + w * 0.5f; q[3] = cy + h * 0.5f;
}

// ===========================================================================
// Host orchestration
// ===========================================================================
static inline int conv_out_dim(int H, int k, int s, int pad, int dil) {
    return (H + 2 * pad - dil * (k - 1) - 1) / s + 1;
}

static void run_conv(hipStream_t st, half_t* wbuf,
                     const float* w, const float* bias,
                     const half_t* in, half_t* outh, float* outf,
                     int Bn, int Cin, int H, int W,
                     int Cout, int k, int stride, int pad, int dil, int relu,
                     int* Ho, int* Wo)
{
    long wn = (long)Cout * Cin * k * k;
    cvt_f32_f16<<<(int)((wn + 255) / 256), 256, 0, st>>>(w, wbuf, wn);
    int Hout = conv_out_dim(H, k, stride, pad, dil);
    int Wout = conv_out_dim(W, k, stride, pad, dil);
    long P = (long)Bn * Hout * Wout;
    dim3 grid((unsigned)((P + 63) / 64), (unsigned)((Cout + 63) / 64));
    conv_wmma_kernel<<<grid, NTHR, 0, st>>>(in, wbuf, bias, outh, outf,
                                            Bn, Cin, H, W, Cout, Hout, Wout,
                                            k, k, stride, pad, dil, relu);
    if (Ho) *Ho = Hout;
    if (Wo) *Wo = Wout;
}

extern "C" void kernel_launch(void* const* d_in, const int* in_sizes, int n_in,
                              void* d_out, int out_size, void* d_ws, size_t ws_size,
                              hipStream_t stream)
{
    (void)in_sizes; (void)n_in; (void)out_size; (void)ws_size;

    int ii = 0;
    const float* x = (const float*)d_in[ii++];
    const float *vw[13], *vb[13];
    for (int l = 0; l < 13; ++l) { vw[l] = (const float*)d_in[ii++]; vb[l] = (const float*)d_in[ii++]; }
    const float* fc6w = (const float*)d_in[ii++]; const float* fc6b = (const float*)d_in[ii++];
    const float* fc7w = (const float*)d_in[ii++]; const float* fc7b = (const float*)d_in[ii++];
    const float* l2s  = (const float*)d_in[ii++];
    const float *ew[8], *eb[8];
    for (int l = 0; l < 8; ++l) { ew[l] = (const float*)d_in[ii++]; eb[l] = (const float*)d_in[ii++]; }
    const float *lw[6], *lbp[6], *cw[6], *cbp[6];
    for (int l = 0; l < 6; ++l) {
        lw[l]  = (const float*)d_in[ii++]; lbp[l] = (const float*)d_in[ii++];
        cw[l]  = (const float*)d_in[ii++]; cbp[l] = (const float*)d_in[ii++];
    }
    const float* priors = (const float*)d_in[ii++];

    // -------- workspace carve --------
    char* wsp = (char*)d_ws;
    size_t off = 0;
    auto carve = [&](size_t bytes) -> char* {
        char* p = wsp + off;
        off += (bytes + 255) & ~(size_t)255;
        return p;
    };
    const long NA = 32L * 64 * 300 * 300;                 // largest activation (elems)
    half_t* bufA = (half_t*)carve((size_t)NA * 2);
    half_t* bufB = (half_t*)carve((size_t)NA * 2);
    half_t* wbuf = (half_t*)carve((size_t)8 * 1024 * 1024 * 2);
    half_t* src0 = (half_t*)carve((size_t)32 * 512  * 38 * 38 * 2);
    half_t* src1 = (half_t*)carve((size_t)32 * 1024 * 19 * 19 * 2);
    half_t* src2 = (half_t*)carve((size_t)32 * 512  * 10 * 10 * 2);
    half_t* src3 = (half_t*)carve((size_t)32 * 256  * 5  * 5  * 2);
    half_t* src4 = (half_t*)carve((size_t)32 * 256  * 3  * 3  * 2);
    half_t* src5 = (half_t*)carve((size_t)32 * 256  * 1  * 1  * 2);
    float*  htmp = (float*)carve((size_t)4 * 1024 * 1024 * 4);
    float*  locb = (float*)carve((size_t)32 * 8732 * 4  * 4);
    float*  confb= (float*)carve((size_t)32 * 8732 * 21 * 4);

    const int Bn = 32;
    auto run_pool = [&](const half_t* in, half_t* out, int C_, int Hin, int Win,
                        int k, int s, int padTL, int Hout, int Wout) {
        long n = (long)Bn * C_ * Hout * Wout;
        maxpool_kernel<<<(int)((n + 255) / 256), 256, 0, stream>>>(
            in, out, Bn, C_, Hin, Win, Hout, Wout, k, s, padTL, padTL);
    };

    // -------- input fp32 -> fp16 --------
    { long n = 32L * 3 * 300 * 300;
      cvt_f32_f16<<<(int)((n + 255) / 256), 256, 0, stream>>>(x, bufA, n); }

    // -------- VGG trunk --------
    run_conv(stream, wbuf, vw[0], vb[0], bufA, bufB, nullptr, Bn, 3,   300, 300, 64,  3, 1, 1, 1, 1, 0, 0);
    run_conv(stream, wbuf, vw[1], vb[1], bufB, bufA, nullptr, Bn, 64,  300, 300, 64,  3, 1, 1, 1, 1, 0, 0);
    run_pool(bufA, bufB, 64, 300, 300, 2, 2, 0, 150, 150);
    run_conv(stream, wbuf, vw[2], vb[2], bufB, bufA, nullptr, Bn, 64,  150, 150, 128, 3, 1, 1, 1, 1, 0, 0);
    run_conv(stream, wbuf, vw[3], vb[3], bufA, bufB, nullptr, Bn, 128, 150, 150, 128, 3, 1, 1, 1, 1, 0, 0);
    run_pool(bufB, bufA, 128, 150, 150, 2, 2, 0, 75, 75);
    run_conv(stream, wbuf, vw[4], vb[4], bufA, bufB, nullptr, Bn, 128, 75, 75, 256, 3, 1, 1, 1, 1, 0, 0);
    run_conv(stream, wbuf, vw[5], vb[5], bufB, bufA, nullptr, Bn, 256, 75, 75, 256, 3, 1, 1, 1, 1, 0, 0);
    run_conv(stream, wbuf, vw[6], vb[6], bufA, bufB, nullptr, Bn, 256, 75, 75, 256, 3, 1, 1, 1, 1, 0, 0);
    run_pool(bufB, bufA, 256, 75, 75, 2, 2, 0, 38, 38);   // ceil-mode: trailing pad
    run_conv(stream, wbuf, vw[7], vb[7], bufA, bufB, nullptr, Bn, 256, 38, 38, 512, 3, 1, 1, 1, 1, 0, 0);
    run_conv(stream, wbuf, vw[8], vb[8], bufB, bufA, nullptr, Bn, 512, 38, 38, 512, 3, 1, 1, 1, 1, 0, 0);
    run_conv(stream, wbuf, vw[9], vb[9], bufA, bufB, nullptr, Bn, 512, 38, 38, 512, 3, 1, 1, 1, 1, 0, 0);
    // conv4_3 is in bufB -> L2Norm -> source 0
    { long n = 32L * 38 * 38;
      l2norm_kernel<<<(int)((n + 255) / 256), 256, 0, stream>>>(bufB, l2s, src0, Bn, 512, 38 * 38); }
    run_pool(bufB, bufA, 512, 38, 38, 2, 2, 0, 19, 19);
    run_conv(stream, wbuf, vw[10], vb[10], bufA, bufB, nullptr, Bn, 512, 19, 19, 512, 3, 1, 1, 1, 1, 0, 0);
    run_conv(stream, wbuf, vw[11], vb[11], bufB, bufA, nullptr, Bn, 512, 19, 19, 512, 3, 1, 1, 1, 1, 0, 0);
    run_conv(stream, wbuf, vw[12], vb[12], bufA, bufB, nullptr, Bn, 512, 19, 19, 512, 3, 1, 1, 1, 1, 0, 0);
    run_pool(bufB, bufA, 512, 19, 19, 3, 1, 1, 19, 19);   // pool5: 3x3 s1 p1
    run_conv(stream, wbuf, fc6w, fc6b, bufA, bufB, nullptr, Bn, 512,  19, 19, 1024, 3, 1, 6, 6, 1, 0, 0); // dilated
    run_conv(stream, wbuf, fc7w, fc7b, bufB, src1, nullptr, Bn, 1024, 19, 19, 1024, 1, 1, 0, 1, 1, 0, 0); // source 1

    // -------- extras --------
    run_conv(stream, wbuf, ew[0], eb[0], src1, bufA, nullptr, Bn, 1024, 19, 19, 256, 1, 1, 0, 1, 1, 0, 0);
    run_conv(stream, wbuf, ew[1], eb[1], bufA, src2, nullptr, Bn, 256,  19, 19, 512, 3, 2, 1, 1, 1, 0, 0); // 10x10, source 2
    run_conv(stream, wbuf, ew[2], eb[2], src2, bufA, nullptr, Bn, 512,  10, 10, 128, 1, 1, 0, 1, 1, 0, 0);
    run_conv(stream, wbuf, ew[3], eb[3], bufA, src3, nullptr, Bn, 128,  10, 10, 256, 3, 2, 1, 1, 1, 0, 0); // 5x5, source 3
    run_conv(stream, wbuf, ew[4], eb[4], src3, bufA, nullptr, Bn, 256,  5,  5,  128, 1, 1, 0, 1, 1, 0, 0);
    run_conv(stream, wbuf, ew[5], eb[5], bufA, src4, nullptr, Bn, 128,  5,  5,  256, 3, 1, 0, 1, 1, 0, 0); // 3x3, source 4
    run_conv(stream, wbuf, ew[6], eb[6], src4, bufA, nullptr, Bn, 256,  3,  3,  128, 1, 1, 0, 1, 1, 0, 0);
    run_conv(stream, wbuf, ew[7], eb[7], bufA, src5, nullptr, Bn, 128,  3,  3,  256, 3, 1, 0, 1, 1, 0, 0); // 1x1, source 5

    // -------- multibox heads --------
    half_t* S[6]   = { src0, src1, src2, src3, src4, src5 };
    int sc[6]      = { 512, 1024, 512, 256, 256, 256 };
    int sh[6]      = { 38, 19, 10, 5, 3, 1 };
    int nbx[6]     = { 4, 6, 6, 6, 4, 4 };
    int poff[6]    = { 0, 5776, 7942, 8542, 8692, 8728 };
    for (int s = 0; s < 6; ++s) {
        // loc head (3x3 pad1, no relu, fp32 out)
        run_conv(stream, wbuf, lw[s], lbp[s], S[s], nullptr, htmp,
                 Bn, sc[s], sh[s], sh[s], nbx[s] * 4, 3, 1, 1, 1, 0, 0, 0);
        long n1 = (long)Bn * sh[s] * sh[s] * nbx[s] * 4;
        scatter_head<<<(int)((n1 + 255) / 256), 256, 0, stream>>>(
            htmp, locb, Bn, sh[s], sh[s], nbx[s], 4, poff[s], 8732);
        // conf head
        run_conv(stream, wbuf, cw[s], cbp[s], S[s], nullptr, htmp,
                 Bn, sc[s], sh[s], sh[s], nbx[s] * 21, 3, 1, 1, 1, 0, 0, 0);
        long n2 = (long)Bn * sh[s] * sh[s] * nbx[s] * 21;
        scatter_head<<<(int)((n2 + 255) / 256), 256, 0, stream>>>(
            htmp, confb, Bn, sh[s], sh[s], nbx[s], 21, poff[s], 8732);
    }

    // -------- decode + softmax into d_out (boxes first, then conf_sm) --------
    float* boxes  = (float*)d_out;                  // [32, 8732, 4]
    float* confsm = boxes + 32L * 8732 * 4;         // [32, 8732, 21]
    { long n = 32L * 8732;
      decode_kernel<<<(int)((n + 255) / 256), 256, 0, stream>>>(locb, priors, boxes, Bn, 8732);
      softmax21_kernel<<<(int)((n + 255) / 256), 256, 0, stream>>>(confb, confsm, n); }
}
